// MHA_56504589746420
// MI455X (gfx1250) — matrix operand order repaired and checked by reading, not compiled, not run on hardware
//
#include <hip/hip_runtime.h>
#include <hip/hip_bf16.h>

// ---------- types ----------
typedef __bf16 bf16;
typedef bf16  v16bf __attribute__((ext_vector_type(16)));
typedef bf16  v8bf  __attribute__((ext_vector_type(8)));
typedef float v8f   __attribute__((ext_vector_type(8)));

#define DEV __device__ __forceinline__

// round-to-nearest-even f32 -> bf16
DEV bf16 f2bf(float f) {
  unsigned u = __builtin_bit_cast(unsigned, f);
  u += 0x7FFFu + ((u >> 16) & 1u);
  unsigned short h = (unsigned short)(u >> 16);
  return __builtin_bit_cast(bf16, h);
}

DEV v16bf cat16(v8bf lo, v8bf hi) {
  return __builtin_shufflevector(lo, hi, 0,1,2,3,4,5,6,7,8,9,10,11,12,13,14,15);
}

// A fragment (16x32 bf16, row-major source, ld in elements).
// lanes 0-15: row m=lane, K {0..7,16..23}; lanes 16-31: row m=lane-16, K {8..15,24..31}.
DEV v16bf load_a_frag(const bf16* A, int lda, int row0, int k0, int lane) {
  const int m  = lane & 15;
  const int kb = (lane >> 4) << 3;              // 0 or 8
  const bf16* p = A + (size_t)(row0 + m) * lda + k0 + kb;
  v8bf lo = *(const v8bf*)(p);                  // K kb .. kb+7
  v8bf hi = *(const v8bf*)(p + 16);             // K kb+16 .. kb+23
  return cat16(lo, hi);
}

// B fragment (32x16 bf16) from Bt where Bt[n][k] is contiguous in k (ldb elements).
// lanes 0-15: col n=lane, K {0..7,16..23}; lanes 16-31: col n=lane-16, K {8..15,24..31}.
DEV v16bf load_b_frag(const bf16* Bt, int ldb, int k0, int n0, int lane) {
  const int n  = lane & 15;
  const int kb = (lane >> 4) << 3;              // 0 or 8
  const bf16* p = Bt + (size_t)(n0 + n) * ldb + k0 + kb;
  v8bf lo = *(const v8bf*)(p);
  v8bf hi = *(const v8bf*)(p + 16);
  return cat16(lo, hi);
}

DEV v8f wmma_bf16(v16bf a, v16bf b, v8f c) {
  return __builtin_amdgcn_wmma_f32_16x16x32_bf16(false, a, false, b, (short)0, c, false, false);
}

// async 16B copy global -> LDS (ASYNCcnt tracked). lds_off = low 32 bits of
// generic shared pointer == wave-relative LDS byte address (ISA 10.2 aperture rules).
DEV void async_copy16(unsigned lds_off, const bf16* g) {
  asm volatile("global_load_async_to_lds_b128 %0, %1, off"
               :: "v"(lds_off), "v"((unsigned long long)(size_t)g) : "memory");
}
DEV void wait_async0() { asm volatile("s_wait_asynccnt 0" ::: "memory"); }

// ---------- conversion kernels ----------
__global__ void cvt_f32_bf16(const float* __restrict__ src, bf16* __restrict__ dst, int n) {
  int i = blockIdx.x * blockDim.x + threadIdx.x;
  if (i < n) dst[i] = f2bf(src[i]);
}

// src is [Kdim, N] row-major f32 -> dst is [N, Kdim] row-major bf16
__global__ void transpose_cvt(const float* __restrict__ src, bf16* __restrict__ dst,
                              int Kdim, int N) {
  int i = blockIdx.x * blockDim.x + threadIdx.x;
  if (i < Kdim * N) {
    int k = i / N, n = i % N;
    dst[(size_t)n * Kdim + k] = f2bf(src[i]);
  }
}

// ---------- QKV projection: C[8192,3072] = X @ Wqkv + b; scatter into Q/K/Vt ----------
// 64x32 tile per wave (4x2 WMMA tiles): 8 WMMAs per 6 fragment loads per k-step.
__global__ __launch_bounds__(128) void qkv_gemm_kernel(
    const bf16* __restrict__ X, const bf16* __restrict__ Wt,
    const float* __restrict__ bias,
    bf16* __restrict__ Q, bf16* __restrict__ K, bf16* __restrict__ Vt) {
  const int lane = threadIdx.x & 31;
  const int wid  = blockIdx.x * 4 + (threadIdx.x >> 5);
  const int NT   = 3072 / 32;                   // 96 col-tiles
  const int m0   = (wid / NT) * 64;
  const int n0   = (wid % NT) * 32;

  v8f c[4][2] = {};
  for (int k0 = 0; k0 < 1024; k0 += 32) {
    v16bf a[4], bb[2];
    #pragma unroll
    for (int i = 0; i < 4; ++i) a[i] = load_a_frag(X, 1024, m0 + 16 * i, k0, lane);
    #pragma unroll
    for (int jn = 0; jn < 2; ++jn) bb[jn] = load_b_frag(Wt, 1024, k0, n0 + 16 * jn, lane);
    #pragma unroll
    for (int i = 0; i < 4; ++i)
      #pragma unroll
      for (int jn = 0; jn < 2; ++jn)
        c[i][jn] = wmma_bf16(a[i], bb[jn], c[i][jn]);
  }

  const int col = lane & 15;
  const int rb  = (lane >> 4) * 8;
  #pragma unroll
  for (int ti = 0; ti < 4; ++ti)
    #pragma unroll
    for (int tj = 0; tj < 2; ++tj)
      #pragma unroll
      for (int r = 0; r < 8; ++r) {
        int gm = m0 + ti * 16 + rb + r;
        int gn = n0 + tj * 16 + col;
        float v = c[ti][tj][r] + bias[gn];
        int sel = gn >> 10;                     // 0=q 1=k 2=v
        int hn  = gn & 1023;
        int h   = hn >> 6, d = hn & 63;
        int b2  = gm >> 11, l = gm & 2047;
        size_t qk = ((size_t)(b2 * 16 + h) * 2048 + l) * 64 + d;
        if (sel == 0)      Q[qk] = f2bf(v * 0.125f);   // fold 1/sqrt(64) into Q
        else if (sel == 1) K[qk] = f2bf(v);
        else Vt[((size_t)(b2 * 16 + h) * 64 + d) * 2048 + l] = f2bf(v);
      }
}

// ---------- flash attention ----------
// Block = 4 waves sharing (b,h); K/V tiles staged once per block into LDS via
// async-to-LDS (the 4 waves previously issued 4x redundant global loads).
// Padded LDS strides: K row 144B (bank step 36), V/P row 80B (bank step 20)
// -> 16 distinct banks across each 16-lane half for fragment reads.
__global__ __launch_bounds__(128) void attn_kernel(
    const bf16* __restrict__ Q, const bf16* __restrict__ K,
    const bf16* __restrict__ Vt, bf16* __restrict__ O) {
  __shared__ __attribute__((aligned(16))) bf16 sK[32 * 72];      // 4608 B
  __shared__ __attribute__((aligned(16))) bf16 sV[64 * 40];      // 5120 B
  __shared__ __attribute__((aligned(16))) bf16 sP[4][16 * 40];   // 5120 B

  const int tid  = threadIdx.x;
  const int lane = tid & 31;
  const int wv   = tid >> 5;
  const int b = blockIdx.z, h = blockIdx.y;
  const int q0 = (blockIdx.x * 4 + wv) * 16;

  const bf16* Qh = Q  + (size_t)(b * 16 + h) * 2048 * 64;
  const bf16* Kh = K  + (size_t)(b * 16 + h) * 2048 * 64;
  const bf16* Vh = Vt + (size_t)(b * 16 + h) * 64 * 2048;

  v16bf aq0 = load_a_frag(Qh, 64, q0, 0, lane);
  v16bf aq1 = load_a_frag(Qh, 64, q0, 32, lane);

  v8f o0 = {}, o1 = {}, o2 = {}, o3 = {};
  float mrow[8], lrow[8];
  #pragma unroll
  for (int r = 0; r < 8; ++r) { mrow[r] = -1e30f; lrow[r] = 0.0f; }

  bf16* Pw = &sP[wv][0];
  const unsigned kbase = (unsigned)(size_t)(&sK[0]);
  const unsigned vbase = (unsigned)(size_t)(&sV[0]);
  const int col = lane & 15;
  const int rb  = (lane >> 4) * 8;

  for (int j = 0; j < 2048; j += 32) {
    // stage K tile [32 keys x 64 d] and V tile [64 d x 32 keys] into LDS:
    // 256 + 256 16B chunks over 128 threads
    #pragma unroll
    for (int t = 0; t < 2; ++t) {
      int id  = tid + t * 128;
      int key = id >> 3, c = id & 7;
      async_copy16(kbase + (unsigned)(key * 144 + c * 16),
                   Kh + (size_t)(j + key) * 64 + c * 8);
    }
    #pragma unroll
    for (int t = 0; t < 2; ++t) {
      int id = tid + t * 128;
      int d  = id >> 2, c = id & 3;
      async_copy16(vbase + (unsigned)(d * 80 + c * 16),
                   Vh + (size_t)d * 2048 + j + c * 8);
    }
    wait_async0();
    __syncthreads();

    // S tiles: 16 queries x 32 keys (Q pre-scaled by 1/8)
    v8f s0 = {}, s1 = {};
    s0 = wmma_bf16(aq0, load_b_frag(sK, 72, 0,  0,  lane), s0);
    s0 = wmma_bf16(aq1, load_b_frag(sK, 72, 32, 0,  lane), s0);
    s1 = wmma_bf16(aq0, load_b_frag(sK, 72, 0,  16, lane), s1);
    s1 = wmma_bf16(aq1, load_b_frag(sK, 72, 32, 16, lane), s1);

    // online softmax: each row lives in one 16-lane half of the wave
    #pragma unroll
    for (int r = 0; r < 8; ++r) {
      float vmax = fmaxf(s0[r], s1[r]);
      vmax = fmaxf(vmax, __shfl_xor(vmax, 1, 32));
      vmax = fmaxf(vmax, __shfl_xor(vmax, 2, 32));
      vmax = fmaxf(vmax, __shfl_xor(vmax, 4, 32));
      vmax = fmaxf(vmax, __shfl_xor(vmax, 8, 32));
      float mn  = fmaxf(mrow[r], vmax);
      float fac = __expf(mrow[r] - mn);
      mrow[r] = mn;
      float p0 = __expf(s0[r] - mn);
      float p1 = __expf(s1[r] - mn);
      float ps = p0 + p1;
      ps += __shfl_xor(ps, 1, 32);
      ps += __shfl_xor(ps, 2, 32);
      ps += __shfl_xor(ps, 4, 32);
      ps += __shfl_xor(ps, 8, 32);
      lrow[r] = lrow[r] * fac + ps;
      o0[r] *= fac; o1[r] *= fac; o2[r] *= fac; o3[r] *= fac;
      // C-layout -> row-major bf16 P tile in LDS (padded stride 40)
      Pw[(rb + r) * 40 + col]      = f2bf(p0);
      Pw[(rb + r) * 40 + 16 + col] = f2bf(p1);
    }
    asm volatile("s_wait_dscnt 0" ::: "memory");
    v16bf ap = load_a_frag(Pw, 40, 0, 0, lane); // re-read P in A-fragment layout
    asm volatile("" ::: "memory");

    o0 = wmma_bf16(ap, load_b_frag(sV, 40, 0, 0,  lane), o0);
    o1 = wmma_bf16(ap, load_b_frag(sV, 40, 0, 16, lane), o1);
    o2 = wmma_bf16(ap, load_b_frag(sV, 40, 0, 32, lane), o2);
    o3 = wmma_bf16(ap, load_b_frag(sV, 40, 0, 48, lane), o3);
    __syncthreads();   // all waves done reading sK/sV before next stage
  }

  #pragma unroll
  for (int r = 0; r < 8; ++r) {
    float inv = 1.0f / lrow[r];
    int l = q0 + rb + r;
    bf16* orow = O + ((size_t)b * 2048 + l) * 1024 + h * 64;
    orow[ 0 + col] = f2bf(o0[r] * inv);
    orow[16 + col] = f2bf(o1[r] * inv);
    orow[32 + col] = f2bf(o2[r] * inv);
    orow[48 + col] = f2bf(o3[r] * inv);
  }
}

// ---------- output projection: out[8192,1024] = O @ Wout + b (f32 out) ----------
__global__ __launch_bounds__(128) void out_gemm_kernel(
    const bf16* __restrict__ A, const bf16* __restrict__ Wt,
    const float* __restrict__ bias, float* __restrict__ out) {
  const int lane = threadIdx.x & 31;
  const int wid  = blockIdx.x * 4 + (threadIdx.x >> 5);
  const int NT   = 1024 / 32;                   // 32 col-tiles
  const int m0   = (wid / NT) * 64;
  const int n0   = (wid % NT) * 32;

  v8f c[4][2] = {};
  for (int k0 = 0; k0 < 1024; k0 += 32) {
    v16bf a[4], bb[2];
    #pragma unroll
    for (int i = 0; i < 4; ++i) a[i] = load_a_frag(A, 1024, m0 + 16 * i, k0, lane);
    #pragma unroll
    for (int jn = 0; jn < 2; ++jn) bb[jn] = load_b_frag(Wt, 1024, k0, n0 + 16 * jn, lane);
    #pragma unroll
    for (int i = 0; i < 4; ++i)
      #pragma unroll
      for (int jn = 0; jn < 2; ++jn)
        c[i][jn] = wmma_bf16(a[i], bb[jn], c[i][jn]);
  }
  const int col = lane & 15;
  const int rb  = (lane >> 4) * 8;
  #pragma unroll
  for (int ti = 0; ti < 4; ++ti)
    #pragma unroll
    for (int tj = 0; tj < 2; ++tj)
      #pragma unroll
      for (int r = 0; r < 8; ++r) {
        int gm = m0 + ti * 16 + rb + r;
        int gn = n0 + tj * 16 + col;
        out[(size_t)gm * 1024 + gn] = c[ti][tj][r] + bias[gn];
      }
}

// ---------- launch ----------
extern "C" void kernel_launch(void* const* d_in, const int* in_sizes, int n_in,
                              void* d_out, int out_size, void* d_ws, size_t ws_size,
                              hipStream_t stream) {
  const float* x     = (const float*)d_in[0];
  const float* w_qkv = (const float*)d_in[1];
  const float* b_qkv = (const float*)d_in[2];
  const float* w_out = (const float*)d_in[3];
  const float* b_out = (const float*)d_in[4];
  float* out = (float*)d_out;

  const size_t M = 8192, Dm = 1024;             // M = B*L
  char* ws = (char*)d_ws;
  bf16* Xb  = (bf16*)ws; ws += M * Dm * sizeof(bf16);
  bf16* Wqt = (bf16*)ws; ws += (size_t)3072 * 1024 * sizeof(bf16);
  bf16* Wot = (bf16*)ws; ws += (size_t)1024 * 1024 * sizeof(bf16);
  bf16* Qb  = (bf16*)ws; ws += M * Dm * sizeof(bf16);   // [B,H,L,64], pre-scaled
  bf16* Kb  = (bf16*)ws; ws += M * Dm * sizeof(bf16);   // [B,H,L,64]
  bf16* Vtb = (bf16*)ws; ws += M * Dm * sizeof(bf16);   // [B,H,64,L]
  bf16* Ob  = (bf16*)ws; ws += M * Dm * sizeof(bf16);   // [B,L,1024]

  cvt_f32_bf16<<<(int)((M * Dm + 255) / 256), 256, 0, stream>>>(x, Xb, (int)(M * Dm));
  transpose_cvt<<<(3072 * 1024 + 255) / 256, 256, 0, stream>>>(w_qkv, Wqt, 1024, 3072);
  transpose_cvt<<<(1024 * 1024 + 255) / 256, 256, 0, stream>>>(w_out, Wot, 1024, 1024);

  qkv_gemm_kernel<<<(128 * 96) / 4, 128, 0, stream>>>(Xb, Wqt, b_qkv, Qb, Kb, Vtb);
  attn_kernel<<<dim3(2048 / 64, 16, 4), 128, 0, stream>>>(Qb, Kb, Vtb, Ob);
  out_gemm_kernel<<<(128 * 32) / 4, 128, 0, stream>>>(Ob, Wot, b_out, out);
}
